// Conv3DSubmModule_20658792694257
// MI455X (gfx1250) — compile-verified, matrix operand order checked by reading, not run on hardware
//
#include <hip/hip_runtime.h>

#define DIMD 64
#define DIMP 64
#define KV   27

typedef __attribute__((ext_vector_type(16))) __bf16       v16b;
typedef __attribute__((ext_vector_type(8)))  float        v8f;
typedef __attribute__((ext_vector_type(4)))  float        f4;
typedef __attribute__((ext_vector_type(4)))  unsigned int u4;

union ABf { v16b v; unsigned u[8]; };
union BBf { v16b v; u4 q[2]; };

__device__ __forceinline__ unsigned pack_bf2(float a, float b) {
#if __has_builtin(__builtin_amdgcn_cvt_pk_bf16_f32)
  typedef __attribute__((ext_vector_type(2))) __bf16 v2b;
  v2b p = __builtin_amdgcn_cvt_pk_bf16_f32(a, b);
  return __builtin_bit_cast(unsigned, p);
#else
  unsigned short lo = __builtin_bit_cast(unsigned short, (__bf16)a);
  unsigned short hi = __builtin_bit_cast(unsigned short, (__bf16)b);
  return (unsigned)lo | ((unsigned)hi << 16);
#endif
}

// Convert weights[k] (64x64 f32, row-major [kk][n]) into bf16 WMMA B-fragments in LDS.
// Fragment group fg = kb*4 + nt (kb: K-block of 32, nt: 16-col N-tile).
// 16-bit B layout: lane half h covers K = 8h+j (j<8) and 16+8h+(j-8) (j>=8),
// lane%16 = column. Interleaved u4 layout (q0 at fg*64+lane, q1 at +32) keeps the
// per-lane ds_load_b128 pair at the 2-way-minimum bank pattern.
__device__ __forceinline__ void stage_weights(const float* __restrict__ w,
                                              int k, int buf, int t, u4* lds4) {
  const int lane = t & 31;
  const int fg   = t >> 5;          // 0..7
  const int kb   = fg >> 2;         // 0..1
  const int nt   = fg & 3;          // 0..3
  const int half = (lane >> 4) & 1;
  const int n    = lane & 15;
  const float* base = w + (size_t)k * (DIMD * DIMP)
                        + (size_t)(kb * 32 + 8 * half) * DIMP
                        + (size_t)(nt * 16 + n);
  unsigned u[8];
#pragma unroll
  for (int p = 0; p < 4; ++p)
    u[p]     = pack_bf2(base[(2 * p) * DIMP],      base[(2 * p + 1) * DIMP]);
#pragma unroll
  for (int p = 0; p < 4; ++p)
    u[4 + p] = pack_bf2(base[(16 + 2 * p) * DIMP], base[(17 + 2 * p) * DIMP]);

  u4 q0, q1;
  q0[0] = u[0]; q0[1] = u[1]; q0[2] = u[2]; q0[3] = u[3];
  q1[0] = u[4]; q1[1] = u[5]; q1[2] = u[6]; q1[3] = u[7];
  const int fi = buf * 512 + fg * 64 + lane;
  lds4[fi]      = q0;
  lds4[fi + 32] = q1;
}

// Gather one feature row (lane's channel slices) and build the two K-block
// A-fragments in WMMA 16-bit A layout.
__device__ __forceinline__ void build_a(const float* __restrict__ feats,
                                        int g, int half, ABf& a0, ABf& a1) {
  f4 q0 = {0,0,0,0}, q1 = {0,0,0,0}, q2 = {0,0,0,0}, q3 = {0,0,0,0};
  f4 q4 = {0,0,0,0}, q5 = {0,0,0,0}, q6 = {0,0,0,0}, q7 = {0,0,0,0};
  if (g >= 0) {
    const f4* p = (const f4*)(feats + (size_t)g * DIMD);
    const int b = 2 * half;
    q0 = p[b];      q1 = p[b + 1];   // chans 8h..8h+7         (Kblk0, j 0..7)
    q2 = p[b + 4];  q3 = p[b + 5];   // chans 16+8h..16+8h+7   (Kblk0, j 8..15)
    q4 = p[b + 8];  q5 = p[b + 9];   // Kblk1, j 0..7
    q6 = p[b + 12]; q7 = p[b + 13];  // Kblk1, j 8..15
  }
  a0.u[0] = pack_bf2(q0.x, q0.y); a0.u[1] = pack_bf2(q0.z, q0.w);
  a0.u[2] = pack_bf2(q1.x, q1.y); a0.u[3] = pack_bf2(q1.z, q1.w);
  a0.u[4] = pack_bf2(q2.x, q2.y); a0.u[5] = pack_bf2(q2.z, q2.w);
  a0.u[6] = pack_bf2(q3.x, q3.y); a0.u[7] = pack_bf2(q3.z, q3.w);
  a1.u[0] = pack_bf2(q4.x, q4.y); a1.u[1] = pack_bf2(q4.z, q4.w);
  a1.u[2] = pack_bf2(q5.x, q5.y); a1.u[3] = pack_bf2(q5.z, q5.w);
  a1.u[4] = pack_bf2(q6.x, q6.y); a1.u[5] = pack_bf2(q6.z, q6.w);
  a1.u[6] = pack_bf2(q7.x, q7.y); a1.u[7] = pack_bf2(q7.z, q7.w);
}

__global__ void __launch_bounds__(256)
subm_conv_wmma(const float* __restrict__ feats,
               const int*   __restrict__ idxp,
               const float* __restrict__ w,
               float*       __restrict__ out,
               int nrows) {
  __shared__ u4 lds4[1024];          // 2 x 8KB double-buffered B fragments

  const int t    = threadIdx.x;
  const int lane = t & 31;
  const int wv   = t >> 5;           // wave id 0..7
  const int half = (lane >> 4) & 1;
  const int ln   = lane & 15;

  // Each wave owns two 16-row M-tiles (32 rows) sharing the same B fragments.
  const long rowBase = (long)blockIdx.x * 256 + (long)wv * 32;
  long r0  = rowBase + ln;
  long r1  = rowBase + 16 + ln;
  long rc0 = r0 < nrows ? r0 : (long)(nrows - 1);
  long rc1 = r1 < nrows ? r1 : (long)(nrows - 1);

  v8f acc[2][4];
#pragma unroll
  for (int mt = 0; mt < 2; ++mt)
#pragma unroll
    for (int nt = 0; nt < 4; ++nt)
      acc[mt][nt] = (v8f){0.f,0.f,0.f,0.f,0.f,0.f,0.f,0.f};

  stage_weights(w, 0, 0, t, lds4);

  int g0 = idxp[rc0 * KV];           // software-pipelined gather indices
  int g1 = idxp[rc1 * KV];
  __syncthreads();

  for (int k = 0; k < KV; ++k) {
    if (k + 1 < KV) stage_weights(w, k + 1, (k + 1) & 1, t, lds4);

    // preload next-k indices and prefetch their feature rows (hide the
    // dependent idx->feats chain; one lane per cacheline-half suffices)
    int n0 = -1, n1 = -1;
    if (k + 1 < KV) {
      n0 = idxp[rc0 * KV + (k + 1)];
      n1 = idxp[rc1 * KV + (k + 1)];
      if (ln == 0 && n0 >= 0) __builtin_prefetch(feats + (size_t)n0 * DIMD + 8 * half, 0, 3);
      if (ln == 0 && n1 >= 0) __builtin_prefetch(feats + (size_t)n1 * DIMD + 8 * half, 0, 3);
    }

    ABf a0t0, a1t0, a0t1, a1t1;
    build_a(feats, g0, half, a0t0, a1t0);
    build_a(feats, g1, half, a0t1, a1t1);

    // 16 WMMAs: 4 N-tiles x 2 K-blocks x 2 M-tiles, B fragments from LDS
    const u4* bb = lds4 + (k & 1) * 512;
#pragma unroll
    for (int nt = 0; nt < 4; ++nt) {
      BBf B0, B1;
      const int f0 = nt * 64 + lane;
      const int f1 = (4 + nt) * 64 + lane;
      B0.q[0] = bb[f0]; B0.q[1] = bb[f0 + 32];
      B1.q[0] = bb[f1]; B1.q[1] = bb[f1 + 32];
      acc[0][nt] = __builtin_amdgcn_wmma_f32_16x16x32_bf16(
          false, a0t0.v, false, B0.v, (short)0, acc[0][nt], false, false);
      acc[1][nt] = __builtin_amdgcn_wmma_f32_16x16x32_bf16(
          false, a0t1.v, false, B0.v, (short)0, acc[1][nt], false, false);
      acc[0][nt] = __builtin_amdgcn_wmma_f32_16x16x32_bf16(
          false, a1t0.v, false, B1.v, (short)0, acc[0][nt], false, false);
      acc[1][nt] = __builtin_amdgcn_wmma_f32_16x16x32_bf16(
          false, a1t1.v, false, B1.v, (short)0, acc[1][nt], false, false);
    }
    g0 = n0;
    g1 = n1;
    __syncthreads();
  }

  // ---- store C/D: VGPR v -> row (v + 8*half), lane%16 -> column
#pragma unroll
  for (int mt = 0; mt < 2; ++mt) {
#pragma unroll
    for (int nt = 0; nt < 4; ++nt) {
#pragma unroll
      for (int v = 0; v < 8; ++v) {
        long row = rowBase + mt * 16 + 8 * half + v;
        if (row < nrows) out[row * DIMP + nt * 16 + ln] = acc[mt][nt][v];
      }
    }
  }
}

extern "C" void kernel_launch(void* const* d_in, const int* in_sizes, int n_in,
                              void* d_out, int out_size, void* d_ws, size_t ws_size,
                              hipStream_t stream) {
  const float* feats = (const float*)d_in[0];
  const int*   idx   = (const int*)d_in[1];
  const float* w     = (const float*)d_in[2];
  float* out = (float*)d_out;

  const int nrows = in_sizes[0] / DIMD;          // N = 200000
  const int grid  = (nrows + 255) / 256;         // 256 rows per block (8 waves x 32)
  subm_conv_wmma<<<grid, 256, 0, stream>>>(feats, idx, w, out, nrows);
}
